// StreamPETRHeadLite_62517543960650
// MI455X (gfx1250) — matrix operand order repaired and checked by reading, not compiled, not run on hardware
//
#include <hip/hip_runtime.h>
#include <hip/hip_bf16.h>

typedef __attribute__((ext_vector_type(16))) _Float16 v16h;
typedef __attribute__((ext_vector_type(8)))  float    v8f;

// Global-address-space pointers: force global_load/global_store (not flat),
// and enable SADDR(uniform base) + 32-bit VADDR offset addressing.
typedef const __attribute__((address_space(1))) float* gcf;
typedef __attribute__((address_space(1))) float*       gf;
__device__ __forceinline__ gcf to_gc(const float* p) { return (gcf)(unsigned long long)p; }
__device__ __forceinline__ gf  to_g (float* p)       { return (gf )(unsigned long long)p; }

#define FLAG_RELU 1
#define FLAG_ADDC 2

__device__ __constant__ float kTwoPi = 6.283185307179586f;

// ---------------------------------------------------------------------------
// Generic strided, batched WMMA GEMM:  C = op(A @ B + bias [+ C])
// A[m,k] at A[m*sAm + k*sAk], B[k,n] at B[k*sBk + n*sBn], C[m,n] at C[m*sCm+n*sCn]
// One wave (32 lanes) computes a 16x64 C tile: A fragment reused across 4 WMMAs.
// K must be a multiple of 32. Bounds handled branchlessly (clamped index +
// value select) so EXEC stays all-ones through the WMMA ops.
// Uniform bases advance per K-step (scalar adds); per-lane offsets are
// loop-invariant 32-bit ints -> global_load with saddr+voffset.
// ---------------------------------------------------------------------------
__global__ __launch_bounds__(32)
void k_gemm_wmma(const float* __restrict__ Ap, const float* __restrict__ Bp,
                 const float* __restrict__ biasp, float* __restrict__ Cp,
                 int M, int N, int K,
                 int sAm, int sAk, int sBk, int sBn, int sCm, int sCn,
                 long long bsA, long long bsB, long long bsBias, long long bsC,
                 int wDiv, int flags)
{
    const int lane  = threadIdx.x;
    const int g     = lane >> 4;        // half-wave select
    const int ln16  = lane & 15;
    const int batch = blockIdx.z;
    const int wb    = batch / wDiv;

    gcf Abase = to_gc(Ap    + (long long)batch * bsA);
    gcf Bbase = to_gc(Bp    + (long long)wb    * bsB);
    gcf Bias  = to_gc(biasp + (long long)wb    * bsBias);
    gf  Cbase = to_g (Cp    + (long long)batch * bsC);

    const int m0 = blockIdx.y * 16;
    const int n0 = blockIdx.x * 64;

    // ---- per-lane loop-invariant element offsets ----
    const int  am   = m0 + ln16;
    const bool amOK = (am < M);
    const int  amC  = amOK ? am : (M - 1);

    int aIdx[8];                         // A 16x32 f16 fragment layout (ISA 7.12.2)
#pragma unroll
    for (int j = 0; j < 8; ++j)
        aIdx[j] = amC * sAm + ((((j >> 2) << 4) + (g << 3) + ((j & 3) << 1)) * sAk);

    bool bOK[4];
    int  bIdx[4][8];                     // B 32x16 f16 fragment layout
#pragma unroll
    for (int t = 0; t < 4; ++t) {
        int bn  = n0 + t * 16 + ln16;
        bOK[t]  = (bn < N);
        int bnC = bOK[t] ? bn : (N - 1);
#pragma unroll
        for (int j = 0; j < 8; ++j)
            bIdx[t][j] = bnC * sBn + ((((j >> 2) << 4) + (g << 3) + ((j & 3) << 1)) * sBk);
    }

    v8f acc[4] = {v8f{}, v8f{}, v8f{}, v8f{}};

    for (int kk = 0; kk < K; kk += 32) {
        // Uniform (scalar) K-step bases; per-lane 32-bit offsets stay fixed.
        gcf Ak0 = Abase + (long long)kk * sAk;
        gcf Ak1 = Ak0 + sAk;
        gcf Bk0 = Bbase + (long long)kk * sBk;
        gcf Bk1 = Bk0 + sBk;

        v16h a;
#pragma unroll
        for (int j = 0; j < 8; ++j) {
            float x0 = Ak0[aIdx[j]];
            float x1 = Ak1[aIdx[j]];
            x0 = amOK ? x0 : 0.0f;
            x1 = amOK ? x1 : 0.0f;
            a[2 * j]     = (_Float16)x0;
            a[2 * j + 1] = (_Float16)x1;
        }
        v16h bf[4];
#pragma unroll
        for (int t = 0; t < 4; ++t) {
#pragma unroll
            for (int j = 0; j < 8; ++j) {
                float x0 = Bk0[bIdx[t][j]];
                float x1 = Bk1[bIdx[t][j]];
                x0 = bOK[t] ? x0 : 0.0f;
                x1 = bOK[t] ? x1 : 0.0f;
                bf[t][2 * j]     = (_Float16)x0;
                bf[t][2 * j + 1] = (_Float16)x1;
            }
        }
#pragma unroll
        for (int t = 0; t < 4; ++t)
            acc[t] = __builtin_amdgcn_wmma_f32_16x16x32_f16(false, a, false, bf[t],
                                                            (short)0, acc[t], false, false);
    }

    // Epilogue: C/D 16x16 f32 layout -> VGPR r holds M = 8*g + r, N = lane%16
#pragma unroll
    for (int t = 0; t < 4; ++t) {
#pragma unroll
        for (int r = 0; r < 8; ++r) {
            int mr = m0 + (g << 3) + r;
            int nc = n0 + t * 16 + ln16;
            if (mr < M && nc < N) {
                float v = acc[t][r] + Bias[nc];
                int idx = mr * sCm + nc * sCn;
                if (flags & FLAG_ADDC) v += Cbase[idx];
                if (flags & FLAG_RELU) v = v > 0.0f ? v : 0.0f;
                Cbase[idx] = v;
            }
        }
    }
}

// ---------------------------------------------------------------------------
// 4x4 matrix inverse (Gauss-Jordan w/ partial pivot), one matrix per thread
// ---------------------------------------------------------------------------
__global__ void k_inv4(const float* __restrict__ src, float* __restrict__ dst, int n)
{
    int t = blockIdx.x * blockDim.x + threadIdx.x;
    if (t >= n) return;
    float a[4][8];
    for (int i = 0; i < 4; ++i)
        for (int j = 0; j < 4; ++j) {
            a[i][j] = src[t * 16 + i * 4 + j];
            a[i][4 + j] = (i == j) ? 1.0f : 0.0f;
        }
    for (int col = 0; col < 4; ++col) {
        int piv = col; float best = fabsf(a[col][col]);
        for (int r = col + 1; r < 4; ++r) { float f = fabsf(a[r][col]); if (f > best) { best = f; piv = r; } }
        if (piv != col)
            for (int j = 0; j < 8; ++j) { float tmp = a[col][j]; a[col][j] = a[piv][j]; a[piv][j] = tmp; }
        float ip = 1.0f / a[col][col];
        for (int j = 0; j < 8; ++j) a[col][j] *= ip;
        for (int r = 0; r < 4; ++r) if (r != col) {
            float f = a[r][col];
            for (int j = 0; j < 8; ++j) a[r][j] -= f * a[col][j];
        }
    }
    for (int i = 0; i < 4; ++i)
        for (int j = 0; j < 4; ++j) dst[t * 16 + i * 4 + j] = a[i][4 + j];
}

// ---------------------------------------------------------------------------
// PETR frustum -> inverse-sigmoid 3D coords (c3ws[bn][d*3+i][h*44+w]) + coords_mask
// ---------------------------------------------------------------------------
__global__ void k_c3(const float* __restrict__ invM, const unsigned char* __restrict__ masks,
                     float* __restrict__ c3, float* __restrict__ cmask_out)
{
    int t = blockIdx.x * blockDim.x + threadIdx.x;   // over 24*16*44
    if (t >= 24 * 16 * 44) return;
    int w = t % 44, h = (t / 44) % 16, bn = t / (16 * 44);
    float Mi[16];
    for (int i = 0; i < 16; ++i) Mi[i] = invM[bn * 16 + i];
    const float cw = (float)w * (704.0f / 44.0f);
    const float ch = (float)h * (256.0f / 16.0f);
    const float bin = (61.2f - 1.0f) / (64.0f * 65.0f);
    const float lo[3]   = { -61.2f, -61.2f, -10.0f };
    const float span[3] = { 122.4f, 122.4f,  20.0f };
    int oob = 0;
    float* base = c3 + (long long)bn * 192 * 704 + h * 44 + w;
    for (int d = 0; d < 64; ++d) {
        float cd = 1.0f + bin * (float)d * ((float)d + 1.0f);
        float sc = fmaxf(cd, 1e-5f);
        float x4 = cw * sc, y4 = ch * sc, z4 = cd;
        float c[3];
        c[0] = Mi[0] * x4 + Mi[1] * y4 + Mi[2]  * z4 + Mi[3];
        c[1] = Mi[4] * x4 + Mi[5] * y4 + Mi[6]  * z4 + Mi[7];
        c[2] = Mi[8] * x4 + Mi[9] * y4 + Mi[10] * z4 + Mi[11];
        for (int i = 0; i < 3; ++i) {
            float nv = (c[i] - lo[i]) / span[i];
            oob += (nv > 1.0f) ? 1 : 0;
            oob += (nv < 0.0f) ? 1 : 0;
            float cc  = fminf(fmaxf(nv, 0.0f), 1.0f);
            float num = fmaxf(cc, 1e-5f);
            float den = fmaxf(1.0f - cc, 1e-5f);
            base[(long long)(d * 3 + i) * 704] = logf(num / den);
        }
    }
    bool cm = (oob > 32);                 // > DEPTH*0.5
    bool mk = (masks[t] != 0);
    cmask_out[t] = (mk || cm) ? 1.0f : 0.0f;
}

// ---------------------------------------------------------------------------
// 2D sine positional encoding -> sine_ws[bn][c(256)][h*44+w]
// ---------------------------------------------------------------------------
__global__ void k_sine(const unsigned char* __restrict__ masks, float* __restrict__ sine)
{
    int t = blockIdx.x * blockDim.x + threadIdx.x;
    if (t >= 24 * 16 * 44) return;
    int w = t % 44, h = (t / 44) % 16, bn = t / 704;
    const unsigned char* mrow = masks + bn * 704;
    float y = 0.0f, yl = 0.0f, x = 0.0f, xl = 0.0f;
    for (int hh = 0; hh < 16; ++hh) { float nm = mrow[hh * 44 + w] ? 0.0f : 1.0f; yl += nm; if (hh <= h) y += nm; }
    for (int ww = 0; ww < 44; ++ww) { float nm = mrow[h * 44 + ww] ? 0.0f : 1.0f; xl += nm; if (ww <= w) x += nm; }
    float yn = y / (yl + 1e-6f) * kTwoPi;
    float xn = x / (xl + 1e-6f) * kTwoPi;
    float* base = sine + (long long)bn * 256 * 704 + h * 44 + w;
    for (int c = 0; c < 128; ++c) {
        float dt = powf(10000.0f, (float)((c >> 1) * 2) * (1.0f / 128.0f));
        float py = yn / dt, px = xn / dt;
        base[(long long)c * 704]         = (c & 1) ? cosf(py) : sinf(py);
        base[(long long)(128 + c) * 704] = (c & 1) ? cosf(px) : sinf(px);
    }
}

// ---------------------------------------------------------------------------
// Query position embedding features: qp[q][384] (order y,x,z)
// ---------------------------------------------------------------------------
__global__ void k_qp(const float* __restrict__ ref, float* __restrict__ qp)
{
    int q = blockIdx.x * blockDim.x + threadIdx.x;
    if (q >= 900) return;
    float coords[3] = { ref[q * 3 + 1], ref[q * 3 + 0], ref[q * 3 + 2] };
    for (int c = 0; c < 3; ++c) {
        float p0 = coords[c] * kTwoPi;
        for (int i = 0; i < 128; ++i) {
            float dt = powf(10000.0f, (float)((i >> 1) * 2) * (1.0f / 128.0f));
            float v  = p0 / dt;
            qp[q * 384 + c * 128 + i] = (i & 1) ? cosf(v) : sinf(v);
        }
    }
}

// ---------------------------------------------------------------------------
// Memory refresh: scale by prev_exists, advance timestamp, write rolled tails
// of new_mem_{e,r,t}, and produce time posemb features tp[b*512+m][128]
// ---------------------------------------------------------------------------
__global__ void k_mem(const float* __restrict__ prevex, const float* __restrict__ memE,
                      const float* __restrict__ memR, const float* __restrict__ memT,
                      float* __restrict__ nmE, float* __restrict__ nmR, float* __restrict__ nmT,
                      float* __restrict__ tp)
{
    int t = blockIdx.x * blockDim.x + threadIdx.x;   // B*512
    if (t >= 4 * 512) return;
    int m = t % 512, b = t / 512;
    float prev = prevex[b];
    float mt = memT[t] * prev + prev;
    float p0 = mt * kTwoPi;
    for (int i = 0; i < 128; ++i) {
        float dt = powf(10000.0f, (float)((i >> 1) * 2) * (1.0f / 128.0f));
        float v  = p0 / dt;
        tp[(long long)t * 128 + i] = (i & 1) ? cosf(v) : sinf(v);
    }
    if (m < 256) {  // positions [256:512] of new memory = old memory [0:256] (scaled)
        long long dst = (long long)b * 512 + 256 + m;
        long long src = t;
        for (int e = 0; e < 256; ++e) nmE[dst * 256 + e] = memE[src * 256 + e] * prev;
        for (int i = 0; i < 3; ++i)   nmR[dst * 3 + i]   = memR[src * 3 + i]   * prev;
        nmT[dst] = mt;
    }
}

// ---------------------------------------------------------------------------
// Row-wise LayerNorm (+ReLU) over D=256, one wave per row, gamma/beta per layer
// ---------------------------------------------------------------------------
__global__ __launch_bounds__(32)
void k_ln_relu(float* __restrict__ X, const float* __restrict__ gamma,
               const float* __restrict__ beta, int rowsPerLayer)
{
    const int row = blockIdx.x;
    const int l   = row / rowsPerLayer;
    float* x = X + (long long)row * 256;
    const float* gg = gamma + (long long)l * 256;
    const float* bb = beta  + (long long)l * 256;
    int lane = threadIdx.x;
    float v[8];
    float s = 0.0f;
    for (int i = 0; i < 8; ++i) { v[i] = x[lane * 8 + i]; s += v[i]; }
    for (int off = 16; off; off >>= 1) s += __shfl_xor(s, off, 32);
    float mean = s * (1.0f / 256.0f);
    float q = 0.0f;
    for (int i = 0; i < 8; ++i) { float d = v[i] - mean; q += d * d; }
    for (int off = 16; off; off >>= 1) q += __shfl_xor(q, off, 32);
    float inv = rsqrtf(q * (1.0f / 256.0f) + 1e-5f);
    for (int i = 0; i < 8; ++i) {
        int c = lane * 8 + i;
        float o = (v[i] - mean) * inv * gg[c] + bb[c];
        x[c] = o > 0.0f ? o : 0.0f;
    }
}

// ---------------------------------------------------------------------------
// Per-batch stable top-256 over max-class logit, gather into rolling memory
// ---------------------------------------------------------------------------
__global__ __launch_bounds__(256)
void k_topk(const float* __restrict__ cls5, const float* __restrict__ reg5,
            const float* __restrict__ dec5,
            float* __restrict__ nmE, float* __restrict__ nmR, float* __restrict__ nmT)
{
    __shared__ float sc[900];
    __shared__ int   sel[256];
    __shared__ float rv[256];
    __shared__ int   ri[256];
    const int b = blockIdx.x, tid = threadIdx.x;
    for (int q = tid; q < 900; q += 256) {
        const float* row = cls5 + ((long long)b * 900 + q) * 10;
        float mx = row[0];
        for (int c = 1; c < 10; ++c) mx = fmaxf(mx, row[c]);
        sc[q] = mx;                       // sigmoid is monotonic; rank by logit
    }
    __syncthreads();
    for (int k = 0; k < 256; ++k) {
        float best = -3.0e38f; int bi = 1 << 20;
        for (int q = tid; q < 900; q += 256) {
            float v = sc[q];
            if (v > best || (v == best && q < bi)) { best = v; bi = q; }
        }
        rv[tid] = best; ri[tid] = bi;
        __syncthreads();
        for (int off = 128; off; off >>= 1) {
            if (tid < off) {
                if (rv[tid + off] > rv[tid] ||
                    (rv[tid + off] == rv[tid] && ri[tid + off] < ri[tid])) {
                    rv[tid] = rv[tid + off]; ri[tid] = ri[tid + off];
                }
            }
            __syncthreads();
        }
        if (tid == 0) { sel[k] = ri[0]; sc[ri[0]] = -3.0e38f; }
        __syncthreads();
    }
    // gather: thread k -> rolling-memory slot k
    const int k = tid;
    const int q = sel[k];
    long long dst = (long long)b * 512 + k;
    const float* drow = dec5 + ((long long)b * 900 + q) * 256;
    for (int e = 0; e < 256; ++e) nmE[dst * 256 + e] = drow[e];
    const float* rrow = reg5 + ((long long)b * 900 + q) * 10;
    const float pmin[3] = { -51.2f, -51.2f, -5.0f };
    const float span[3] = { 102.4f, 102.4f,  8.0f };
    for (int i = 0; i < 3; ++i) {
        float v = (rrow[i] - pmin[i]) / fmaxf(span[i], 1e-5f);
        nmR[dst * 3 + i] = fminf(fmaxf(v, 0.0f), 1.0f);
    }
    nmT[dst] = 0.0f;
}

// ---------------------------------------------------------------------------
extern "C" void kernel_launch(void* const* d_in, const int* in_sizes, int n_in,
                              void* d_out, int out_size, void* d_ws, size_t ws_size,
                              hipStream_t stream)
{
    (void)in_sizes; (void)n_in; (void)out_size; (void)ws_size;
    const float* feat   = (const float*)d_in[0];
    const float* l2i    = (const float*)d_in[1];
    const float* prevex = (const float*)d_in[2];
    const float* memE   = (const float*)d_in[3];
    const float* memR   = (const float*)d_in[4];
    const float* memT   = (const float*)d_in[5];
    const float* outsd  = (const float*)d_in[6];
    const float* refp   = (const float*)d_in[7];
    const float* ip_w   = (const float*)d_in[8];
    const float* ip_b   = (const float*)d_in[9];
    const float* pe_w1  = (const float*)d_in[10];
    const float* pe_b1  = (const float*)d_in[11];
    const float* pe_w2  = (const float*)d_in[12];
    const float* pe_b2  = (const float*)d_in[13];
    const float* ap_w1  = (const float*)d_in[14];
    const float* ap_b1  = (const float*)d_in[15];
    const float* ap_w2  = (const float*)d_in[16];
    const float* ap_b2  = (const float*)d_in[17];
    const float* qe_w1  = (const float*)d_in[18];
    const float* qe_b1  = (const float*)d_in[19];
    const float* qe_w2  = (const float*)d_in[20];
    const float* qe_b2  = (const float*)d_in[21];
    const float* te_w1  = (const float*)d_in[22];
    const float* te_b1  = (const float*)d_in[23];
    const float* te_w2  = (const float*)d_in[24];
    const float* te_b2  = (const float*)d_in[25];
    const float* cls_w1 = (const float*)d_in[26];
    const float* cls_b1 = (const float*)d_in[27];
    const float* cls_g1 = (const float*)d_in[28];
    const float* cls_be1= (const float*)d_in[29];
    const float* cls_w2 = (const float*)d_in[30];
    const float* cls_b2 = (const float*)d_in[31];
    const float* cls_g2 = (const float*)d_in[32];
    const float* cls_be2= (const float*)d_in[33];
    const float* cls_w3 = (const float*)d_in[34];
    const float* cls_b3 = (const float*)d_in[35];
    const float* reg_w1 = (const float*)d_in[36];
    const float* reg_b1 = (const float*)d_in[37];
    const float* reg_w2 = (const float*)d_in[38];
    const float* reg_b2 = (const float*)d_in[39];
    const float* reg_w3 = (const float*)d_in[40];
    const float* reg_b3 = (const float*)d_in[41];
    const unsigned char* masks = (const unsigned char*)d_in[42];

    // Output layout (flat, return order)
    float* out   = (float*)d_out;
    float* o_cls = out;                     // [6,4,900,10]  216000
    float* o_reg = o_cls + 216000;          // [6,4,900,10]  216000
    float* o_me  = o_reg + 216000;          // [4,512,256]   524288
    float* o_mr  = o_me  + 524288;          // [4,512,3]     6144
    float* o_mt  = o_mr  + 6144;            // [4,512,1]     2048
    float* o_pos = o_mt  + 2048;            // [4,6,256,16,44] 4325376
    float* o_x   = o_pos + 4325376;         // same          4325376
    float* o_qp  = o_x   + 4325376;         // [900,256]     230400
    float* o_te  = o_qp  + 230400;          // [4,512,256]   524288
    float* o_cm  = o_te  + 524288;          // [4,6,16,44]   16896

    // Workspace layout (phased reuse, peak ~46 MB)
    float* ws     = (float*)d_ws;
    float* w_inv  = ws;                     // 24*16
    float* w_big  = ws + 512;
    float* w_c3   = w_big;                  // 24*192*704 = 3244032
    float* w_peh  = w_big + 3244032;        // 24*704*512 = 8650752
    float* w_sine = w_big;                  // 24*256*704 = 4325376 (after c3 phase)
    float* w_aph  = w_big + 4325376;        // 24*704*256 = 4325376
    float* w_qp   = w_big;                  // 900*384 (after pos phase)
    float* w_qh   = w_qp + 345600;          // 900*256
    float* w_tp   = w_qh + 230400;          // 2048*128
    float* w_th   = w_tp + 262144;          // 2048*256
    float* w_h1   = w_big;                  // 21600*256 (after query/time phase)
    float* w_h2   = w_h1 + 5529600;         // 21600*256

    #define GEMM(A,B,BI,C,M,N,K,sAm,sAk,sBk,sBn,sCm,sCn,bsA,bsB,bsBi,bsC,wDiv,flags,nb) \
        k_gemm_wmma<<<dim3(((N)+63)/64, ((M)+15)/16, (nb)), dim3(32), 0, stream>>>( \
            (A),(B),(BI),(C),(M),(N),(K), \
            (int)(sAm),(int)(sAk),(int)(sBk),(int)(sBn),(int)(sCm),(int)(sCn), \
            (long long)(bsA),(long long)(bsB),(long long)(bsBi),(long long)(bsC),(wDiv),(flags))

    // ---- geometry + 3D PE pipeline ----
    k_inv4<<<1, 32, 0, stream>>>(l2i, w_inv, 24);
    k_c3<<<(16896 + 255) / 256, 256, 0, stream>>>(w_inv, masks, w_c3, o_cm);
    // pe1: [704 x 192] @ [192 x 512] + b, ReLU   (A channel-major)
    GEMM(w_c3, pe_w1, pe_b1, w_peh, 704, 512, 192, 1, 704, 512, 1, 512, 1,
         192LL * 704, 0, 0, 704LL * 512, 1, FLAG_RELU, 24);
    // pe2: [704 x 512] @ [512 x 256] + b -> pos_embed (channel-major out)
    GEMM(w_peh, pe_w2, pe_b2, o_pos, 704, 256, 512, 512, 1, 256, 1, 1, 704,
         704LL * 512, 0, 0, 256LL * 704, 1, 0, 24);
    // ---- 2D sine PE + adapt_pos3d ----
    k_sine<<<(16896 + 255) / 256, 256, 0, stream>>>(masks, w_sine);
    GEMM(w_sine, ap_w1, ap_b1, w_aph, 704, 256, 256, 1, 704, 256, 1, 256, 1,
         256LL * 704, 0, 0, 704LL * 256, 1, FLAG_RELU, 24);
    GEMM(w_aph, ap_w2, ap_b2, o_pos, 704, 256, 256, 256, 1, 256, 1, 1, 704,
         704LL * 256, 0, 0, 256LL * 704, 1, FLAG_ADDC, 24);
    // ---- input projection 1x1 conv ----
    GEMM(feat, ip_w, ip_b, o_x, 704, 256, 256, 1, 704, 256, 1, 1, 704,
         256LL * 704, 0, 0, 256LL * 704, 1, 0, 24);
    // ---- query pos MLP ----
    k_qp<<<(900 + 63) / 64, 64, 0, stream>>>(refp, w_qp);
    GEMM(w_qp, qe_w1, qe_b1, w_qh, 900, 256, 384, 384, 1, 256, 1, 256, 1,
         0, 0, 0, 0, 1, FLAG_RELU, 1);
    GEMM(w_qh, qe_w2, qe_b2, o_qp, 900, 256, 256, 256, 1, 256, 1, 256, 1,
         0, 0, 0, 0, 1, 0, 1);
    // ---- memory refresh + time embedding MLP ----
    k_mem<<<(2048 + 255) / 256, 256, 0, stream>>>(prevex, memE, memR, memT,
                                                  o_me, o_mr, o_mt, w_tp);
    GEMM(w_tp, te_w1, te_b1, w_th, 2048, 256, 128, 128, 1, 256, 1, 256, 1,
         0, 0, 0, 0, 1, FLAG_RELU, 1);
    GEMM(w_th, te_w2, te_b2, o_te, 2048, 256, 256, 256, 1, 256, 1, 256, 1,
         0, 0, 0, 0, 1, 0, 1);
    // ---- classification branch: (Linear -> LN -> ReLU) x2 -> Linear ----
    GEMM(outsd, cls_w1, cls_b1, w_h1, 900, 256, 256, 256, 1, 256, 1, 256, 1,
         900LL * 256, 65536, 256, 900LL * 256, 4, 0, 24);
    k_ln_relu<<<21600, 32, 0, stream>>>(w_h1, cls_g1, cls_be1, 3600);
    GEMM(w_h1, cls_w2, cls_b2, w_h2, 900, 256, 256, 256, 1, 256, 1, 256, 1,
         900LL * 256, 65536, 256, 900LL * 256, 4, 0, 24);
    k_ln_relu<<<21600, 32, 0, stream>>>(w_h2, cls_g2, cls_be2, 3600);
    GEMM(w_h2, cls_w3, cls_b3, o_cls, 900, 10, 256, 256, 1, 10, 1, 10, 1,
         900LL * 256, 2560, 10, 9000, 4, 0, 24);
    // ---- regression branch: (Linear -> ReLU) x2 -> Linear ----
    GEMM(outsd, reg_w1, reg_b1, w_h1, 900, 256, 256, 256, 1, 256, 1, 256, 1,
         900LL * 256, 65536, 256, 900LL * 256, 4, FLAG_RELU, 24);
    GEMM(w_h1, reg_w2, reg_b2, w_h2, 900, 256, 256, 256, 1, 256, 1, 256, 1,
         900LL * 256, 65536, 256, 900LL * 256, 4, FLAG_RELU, 24);
    GEMM(w_h2, reg_w3, reg_b3, o_reg, 900, 10, 256, 256, 1, 10, 1, 10, 1,
         900LL * 256, 2560, 10, 9000, 4, 0, 24);
    // ---- top-k memory update from last decoder layer ----
    k_topk<<<4, 256, 0, stream>>>(o_cls + 5LL * 36000, o_reg + 5LL * 36000,
                                  outsd + 5LL * 4 * 900 * 256, o_me, o_mr, o_mt);
    #undef GEMM
}
